// BlockSparseAttention_13932873908463
// MI455X (gfx1250) — matrix operand order repaired and checked by reading, not compiled, not run on hardware
//
#include <hip/hip_runtime.h>

// ---------------------------------------------------------------------------
// BlockSparseAttention on gfx1250 (MI455X): bf16 WMMA + async global->LDS.
// ---------------------------------------------------------------------------

#define SEQ   2048
#define HID   2048
#define NHEAD 16
#define HDIM  128
#define BSZ   64
#define NBLK  32   // SEQ / BSZ

typedef __bf16 bf16;
typedef __attribute__((ext_vector_type(2)))  __bf16 v2bf;
typedef __attribute__((ext_vector_type(8)))  __bf16 v8bf;
typedef __attribute__((ext_vector_type(16))) __bf16 v16bf;
typedef __attribute__((ext_vector_type(8)))  float  v8f;

__device__ __forceinline__ v8f vzero8() {
  v8f z;
#pragma unroll
  for (int i = 0; i < 8; i++) z[i] = 0.0f;
  return z;
}

// Async global->LDS copy (CDNA5 GLOBAL_LOAD_ASYNC_TO_LDS_B128, tracked with
// ASYNCcnt; VDST VGPR supplies the per-lane LDS byte address, §15.18.3).
__device__ __forceinline__ void async_ld128(const bf16* lds_dst, const bf16* gsrc) {
  unsigned off = (unsigned)(size_t)lds_dst;   // low 32 bits = LDS offset
  asm volatile("global_load_async_to_lds_b128 %0, %1, off"
               :: "v"(off), "v"(gsrc) : "memory");
}
__device__ __forceinline__ void wait_async0() {
  asm volatile("s_wait_asynccnt 0x0" ::: "memory");
}

// A-fragment (16x32 bf16, MxK). ISA 7.12.2: lanes 0-15 row M=lane, K 0..7 in
// v0..v3 and K 16..23 in v4..v7; lanes 16-31 same rows, K 8..15 / 24..31.
__device__ __forceinline__ v16bf load_frag_a(const bf16* base, int ld) {
  const int lane  = threadIdx.x & 31;
  const int row   = lane & 15;
  const int khalf = (lane >> 4) << 3;  // 0 or 8
  const bf16* p0 = base + row * ld + khalf;
  const bf16* p1 = p0 + 16;
  v8bf lo = *(const v8bf*)p0;
  v8bf hi = *(const v8bf*)p1;
  v16bf f;
#pragma unroll
  for (int i = 0; i < 8; i++) { f[i] = lo[i]; f[8 + i] = hi[i]; }
  return f;
}

// B-fragment (32x16 bf16, KxN) from a column-major (transposed) LDS tile:
// baseT[n][k].  Lanes 0-15: col n=lane, K 0..7,16..23; lanes 16-31: K 8..15,24..31.
__device__ __forceinline__ v16bf load_frag_b(const bf16* baseT, int ld) {
  const int lane  = threadIdx.x & 31;
  const int n     = lane & 15;
  const int khalf = (lane >> 4) << 3;  // 0 or 8
  const bf16* p = baseT + n * ld + khalf;
  v8bf lo = *(const v8bf*)p;
  v8bf hi = *(const v8bf*)(p + 16);
  v16bf f;
#pragma unroll
  for (int i = 0; i < 8; i++) { f[i] = lo[i]; f[8 + i] = hi[i]; }
  return f;
}

__device__ __forceinline__ v8f wmma_bf16(v16bf a, v16bf b, v8f c) {
  return __builtin_amdgcn_wmma_f32_16x16x32_bf16(false, a, false, b,
                                                 (short)0, c, false, false);
}

// ---------------------------------------------------------------------------
// fp32 -> bf16 convert
// ---------------------------------------------------------------------------
__global__ __launch_bounds__(256) void cvt_bf16_kernel(const float* __restrict__ in,
                                                       bf16* __restrict__ out, int n) {
  int i = (blockIdx.x * 256 + threadIdx.x) * 4;
  if (i < n) {
    float4 v = *(const float4*)(in + i);
    out[i + 0] = (bf16)v.x; out[i + 1] = (bf16)v.y;
    out[i + 2] = (bf16)v.z; out[i + 3] = (bf16)v.w;
  }
}

// ---------------------------------------------------------------------------
// GEMM: C[M,N] = A[M,K] * B[K,N], bf16 in, f32 accumulate.
// 128x128 block tile, K-step 32, 8 waves; wave -> 32x64 sub-tile.
// Double-buffered pipeline: A tiles stream in via async-to-LDS while the
// current K-step's WMMAs run; B tiles double-buffered through VGPRs (needed
// for the transpose).  OUTF32 selects f32 vs bf16 output at compile time.
// ---------------------------------------------------------------------------
template <bool OUTF32>
__global__ __launch_bounds__(256) void gemm_bf16_kernel(
    const bf16* __restrict__ A, const bf16* __restrict__ Bm,
    void* __restrict__ Cv, int M, int N, int K) {
  __shared__ bf16 As0[128][40], As1[128][40];   // 80B rows (16B aligned)
  __shared__ bf16 Bt0[128][40], Bt1[128][40];   // transposed B tiles

  const int tid  = threadIdx.x;
  const int wave = tid >> 5;
  const int lane = tid & 31;
  const int m0 = blockIdx.y * 128;
  const int n0 = blockIdx.x * 128;
  const int wr = wave >> 1;      // 0..3 -> row strip of 32
  const int wc = wave & 1;       // 0..1 -> col strip of 64

  v8f acc[2][4];
#pragma unroll
  for (int i = 0; i < 2; i++)
#pragma unroll
    for (int j = 0; j < 4; j++) acc[i][j] = vzero8();

  // Per-thread cooperative-load coordinates (fixed across K-steps).
  const int ar = tid >> 2, ac8 = (tid & 3) * 8;          // A: 128 rows x 32 cols
  const int bk2 = (tid >> 4) * 2, bc8 = (tid & 15) * 8;  // B: 16 row-pairs x 16 col-grps

  auto issueA = [&](int k0, bf16 (*As)[40]) {
#pragma unroll
    for (int i = 0; i < 2; i++) {
      int r = ar + i * 64;
      async_ld128(&As[r][ac8], &A[(size_t)(m0 + r) * K + k0 + ac8]);
    }
  };
  auto loadB = [&](int k0, v8bf& b0, v8bf& b1) {
    b0 = *(const v8bf*)&Bm[(size_t)(k0 + bk2 + 0) * N + n0 + bc8];
    b1 = *(const v8bf*)&Bm[(size_t)(k0 + bk2 + 1) * N + n0 + bc8];
  };
  auto storeB = [&](bf16 (*Bt)[40], v8bf b0, v8bf b1) {
#pragma unroll
    for (int j = 0; j < 8; j++) {
      v2bf t; t[0] = b0[j]; t[1] = b1[j];
      *(v2bf*)&Bt[bc8 + j][bk2] = t;
    }
  };
  auto compute = [&](bf16 (*As)[40], bf16 (*Bt)[40]) {
    v16bf a0 = load_frag_a(&As[wr * 32 + 0][0], 40);
    v16bf a1 = load_frag_a(&As[wr * 32 + 16][0], 40);
#pragma unroll
    for (int nt = 0; nt < 4; nt++) {
      v16bf b = load_frag_b(&Bt[wc * 64 + nt * 16][0], 40);
      acc[0][nt] = wmma_bf16(a0, b, acc[0][nt]);
      acc[1][nt] = wmma_bf16(a1, b, acc[1][nt]);
    }
  };

  // Prologue: stage K-step 0 into buffer 0.
  issueA(0, As0);
  {
    v8bf b0, b1;
    loadB(0, b0, b1);
    storeB(Bt0, b0, b1);
  }
  wait_async0();
  __syncthreads();

  for (int k0 = 0; k0 < K; k0 += 64) {
    // Phase A: compute buf0, stage k0+32 -> buf1 (always in range: K%64==0).
    {
      issueA(k0 + 32, As1);
      v8bf b0, b1;
      loadB(k0 + 32, b0, b1);
      if (k0 + 96 < K)
        __builtin_prefetch(&Bm[(size_t)(k0 + 96 + bk2) * N + n0 + bc8], 0, 3);
      compute(As0, Bt0);
      storeB(Bt1, b0, b1);
      wait_async0();
      __syncthreads();
    }
    // Phase B: compute buf1, stage k0+64 -> buf0.
    {
      const bool more = (k0 + 64) < K;
      if (more) {
        issueA(k0 + 64, As0);
        v8bf b0, b1;
        loadB(k0 + 64, b0, b1);
        compute(As1, Bt1);
        storeB(Bt0, b0, b1);
      } else {
        compute(As1, Bt1);
      }
      wait_async0();
      __syncthreads();
    }
  }

  const int col  = lane & 15;
  const int roff = (lane >> 4) ? 8 : 0;
  const size_t cbase = (size_t)(m0 + wr * 32 + roff) * N + (n0 + wc * 64 + col);
  if constexpr (OUTF32) {
    float* C = (float*)Cv;
#pragma unroll
    for (int mt = 0; mt < 2; mt++)
#pragma unroll
      for (int nt = 0; nt < 4; nt++) {
        float* pC = C + cbase + (size_t)(mt * 16) * N + nt * 16;
#pragma unroll
        for (int j = 0; j < 8; j++) { *pC = acc[mt][nt][j]; pC += N; }
      }
  } else {
    bf16* C = (bf16*)Cv;
#pragma unroll
    for (int mt = 0; mt < 2; mt++)
#pragma unroll
      for (int nt = 0; nt < 4; nt++) {
        bf16* pC = C + cbase + (size_t)(mt * 16) * N + nt * 16;
#pragma unroll
        for (int j = 0; j < 8; j++) { *pC = (bf16)acc[mt][nt][j]; pC += N; }
      }
  }
}

// ---------------------------------------------------------------------------
// RoPE applied in place to Q and K (bf16 [SEQ, NHEAD*HDIM]).
// ---------------------------------------------------------------------------
__global__ __launch_bounds__(256) void rope_kernel(bf16* __restrict__ Q,
                                                   bf16* __restrict__ K) {
  int id = blockIdx.x * 256 + threadIdx.x;        // SEQ*NHEAD*64 threads
  int s = id >> 10;                               // / (NHEAD*64)
  int rem = id & 1023;
  int h = rem >> 6;
  int j = rem & 63;
  float invf = __powf(10000.0f, -(float)(2 * j) / 128.0f);
  float ang = (float)s * invf;
  float sn, cs;
  __sincosf(ang, &sn, &cs);
  size_t base = (size_t)s * HID + h * HDIM + j;
  float q0 = (float)Q[base], q1 = (float)Q[base + 64];
  Q[base]      = (bf16)(q0 * cs - q1 * sn);
  Q[base + 64] = (bf16)(q1 * cs + q0 * sn);
  float k0 = (float)K[base], k1 = (float)K[base + 64];
  K[base]      = (bf16)(k0 * cs - k1 * sn);
  K[base + 64] = (bf16)(k1 * cs + k0 * sn);
}

// ---------------------------------------------------------------------------
// Per-head top-512 threshold of 1024 block scores (rank by counting), then
// emit the 0/1 block mask.  One workgroup per head.
// ---------------------------------------------------------------------------
__global__ __launch_bounds__(256) void mask_kernel(const float* __restrict__ sp,
                                                   unsigned char* __restrict__ mask) {
  int h = blockIdx.x, tid = threadIdx.x;
  __shared__ float vals[1024];
  __shared__ float thr_s;
#pragma unroll
  for (int i = 0; i < 4; i++) vals[tid + i * 256] = sp[h * 1024 + tid + i * 256];
  __syncthreads();
#pragma unroll
  for (int i = 0; i < 4; i++) {
    float x = vals[tid + i * 256];
    int cnt = 0;
    for (int j = 0; j < 1024; j++) cnt += (vals[j] > x) ? 1 : 0;
    if (cnt == 511) thr_s = x;   // 512th largest value (kk = 512)
  }
  __syncthreads();
  float thr = thr_s;
#pragma unroll
  for (int i = 0; i < 4; i++) {
    int e = tid + i * 256;
    mask[h * 1024 + e] = vals[e] > thr ? 1 : 0;
  }
}

// ---------------------------------------------------------------------------
// Per (head, key-block) V column sums: vbsum[h][kb][d] = sum_r V[kb*64+r][h,d]
// ---------------------------------------------------------------------------
__global__ __launch_bounds__(128) void vbsum_kernel(const bf16* __restrict__ V,
                                                    float* __restrict__ vbsum) {
  int blk = blockIdx.x;            // 16*32
  int h = blk >> 5, kb = blk & 31, d = threadIdx.x;
  const bf16* base = V + (size_t)(kb * BSZ) * HID + h * HDIM + d;
  float s = 0.0f;
  for (int r = 0; r < BSZ; r++) s += (float)base[(size_t)r * HID];
  vbsum[(size_t)blk * HDIM + d] = s;
}

// ---------------------------------------------------------------------------
// Block-sparse flash attention.  One workgroup per (head, query-block):
// streams selected key blocks with online softmax; masked (zero-score) blocks
// are folded in at the end via precomputed V block sums.  Q/K tiles arrive by
// async global->LDS; V is transposed through VGPRs.
// ---------------------------------------------------------------------------
__global__ __launch_bounds__(256) void attn_kernel(
    const bf16* __restrict__ Qg, const bf16* __restrict__ Kg,
    const bf16* __restrict__ Vg, const unsigned char* __restrict__ mask,
    const float* __restrict__ vbsum, bf16* __restrict__ Og) {
  const int qb = blockIdx.x, h = blockIdx.y;
  const int tid = threadIdx.x, wave = tid >> 5, lane = tid & 31;
  const int col = lane & 15, roff = (lane >> 4) ? 8 : 0;

  __shared__ bf16 Qs[64][136];
  __shared__ bf16 Ks[64][136];
  __shared__ bf16 Vts[128][72];   // V transposed: [dim][key]
  __shared__ float Ss[64][65];
  __shared__ bf16 Ps[64][72];
  __shared__ float mrow[64], lrow[64], arow[64], prow[64];
  __shared__ float vrest[128];
  __shared__ int sel[NBLK], selflag[NBLK], nsel_s;

  if (tid == 0) {
    int n = 0;
    for (int kb = 0; kb < NBLK; kb++) {
      int f = mask[(h * NBLK + qb) * NBLK + kb];
      selflag[kb] = f;
      if (f) sel[n++] = kb;
    }
    nsel_s = n;
  }
  if (tid < 64) { mrow[tid] = -1e30f; lrow[tid] = 0.0f; }

  const bf16* Qbase = Qg + (size_t)(qb * BSZ) * HID + h * HDIM;
#pragma unroll
  for (int i = 0; i < 4; i++) {
    int idx = tid + i * 256;
    int r = idx >> 4, c8 = (idx & 15) * 8;
    async_ld128(&Qs[r][c8], &Qbase[(size_t)r * HID + c8]);
  }
  wait_async0();
  __syncthreads();
  const int nsel = nsel_s;

  v8f acc[4];
#pragma unroll
  for (int t = 0; t < 4; t++) acc[t] = vzero8();

  const float scale = 0.08838834764831845f;   // 1/sqrt(128)

  for (int si = 0; si < nsel; si++) {
    int kb = sel[si];
    const bf16* Kbase = Kg + (size_t)(kb * BSZ) * HID + h * HDIM;
    const bf16* Vbase = Vg + (size_t)(kb * BSZ) * HID + h * HDIM;
    // K rows async-copied straight; V transposed with packed b32 stores.
#pragma unroll
    for (int i = 0; i < 2; i++) {
      int idx = tid + i * 256;
      int r2 = (idx >> 4) * 2;
      int c8 = (idx & 15) * 8;
      async_ld128(&Ks[r2 + 0][c8], &Kbase[(size_t)(r2 + 0) * HID + c8]);
      async_ld128(&Ks[r2 + 1][c8], &Kbase[(size_t)(r2 + 1) * HID + c8]);
      v8bf v0 = *(const v8bf*)&Vbase[(size_t)(r2 + 0) * HID + c8];
      v8bf v1 = *(const v8bf*)&Vbase[(size_t)(r2 + 1) * HID + c8];
#pragma unroll
      for (int j = 0; j < 8; j++) {
        v2bf t; t[0] = v0[j]; t[1] = v1[j];
        *(v2bf*)&Vts[c8 + j][r2] = t;
      }
    }
    wait_async0();
    __syncthreads();

    // S = Q K^T * scale : 4x4 tiles of 16x16; each wave does 2 tiles.
#pragma unroll
    for (int tt = 0; tt < 2; tt++) {
      int t = wave + tt * 8;
      int mt = t >> 2, nt = t & 3;
      v8f s = vzero8();
#pragma unroll
      for (int kk = 0; kk < HDIM; kk += 32) {
        v16bf a = load_frag_a(&Qs[mt * 16][kk], 136);
        v16bf b = load_frag_b(&Ks[nt * 16][kk], 136);  // B = K^T: Kt[n][k]=K[n][k]
        s = wmma_bf16(a, b, s);
      }
#pragma unroll
      for (int j = 0; j < 8; j++)
        Ss[mt * 16 + j + roff][nt * 16 + col] = s[j] * scale;
    }
    __syncthreads();

    // Online softmax row update (one thread per query row).
    if (tid < 64) {
      int r = tid;
      float bmax = -1e30f;
      for (int c = 0; c < 64; c++) bmax = fmaxf(bmax, Ss[r][c]);
      float nm = fmaxf(mrow[r], bmax);
      float al = __expf(mrow[r] - nm);
      float sum = 0.0f;
      for (int c = 0; c < 64; c++) {
        float p = __expf(Ss[r][c] - nm);
        Ps[r][c] = (bf16)p;
        sum += p;
      }
      lrow[r] = lrow[r] * al + sum;
      mrow[r] = nm;
      arow[r] = al;
    }
    __syncthreads();

    // Rescale accumulators, then O += P * V.  Wave owns dim strip wave*16.
#pragma unroll
    for (int mt = 0; mt < 4; mt++) {
#pragma unroll
      for (int j = 0; j < 8; j++) acc[mt][j] *= arow[mt * 16 + j + roff];
#pragma unroll
      for (int kk = 0; kk < BSZ; kk += 32) {
        v16bf a = load_frag_a(&Ps[mt * 16][kk], 72);
        v16bf b = load_frag_b(&Vts[wave * 16][kk], 72);
        acc[mt] = wmma_bf16(a, b, acc[mt]);
      }
    }
    __syncthreads();
  }

  // Fold in masked (score==0) positions: prob p0 = exp(-m)/Z each, and their
  // PV contribution is p0 * (sum of V over unselected blocks).
  if (tid < 128) {
    float s = 0.0f;
    for (int kb = 0; kb < NBLK; kb++)
      if (!selflag[kb]) s += vbsum[((size_t)(h * NBLK + kb)) * HDIM + tid];
    vrest[tid] = s;
  }
  __syncthreads();
  if (tid < 64) {
    int r = tid;
    float nm = fmaxf(mrow[r], 0.0f);
    float al = __expf(mrow[r] - nm);
    float p0 = __expf(0.0f - nm);
    float nmasked = (float)((NBLK - nsel) * BSZ);
    lrow[r] = lrow[r] * al + nmasked * p0;
    arow[r] = al;
    prow[r] = p0;
  }
  __syncthreads();

  {
    int d = wave * 16 + col;
    float vr = vrest[d];
    bf16* pO0 = Og + (size_t)(qb * BSZ + roff) * HID + h * HDIM + d;
#pragma unroll
    for (int mt = 0; mt < 4; mt++) {
      bf16* pO = pO0 + (size_t)(mt * 16) * HID;
#pragma unroll
      for (int j = 0; j < 8; j++) {
        int r = mt * 16 + j + roff;
        float o = (acc[mt][j] * arow[r] + prow[r] * vr) / lrow[r];
        *pO = (bf16)o;
        pO += HID;
      }
    }
  }
}

// ---------------------------------------------------------------------------
// Orchestration
// ---------------------------------------------------------------------------
extern "C" void kernel_launch(void* const* d_in, const int* in_sizes, int n_in,
                              void* d_out, int out_size, void* d_ws, size_t ws_size,
                              hipStream_t stream) {
  (void)in_sizes; (void)n_in; (void)out_size; (void)ws_size;
  const float* hs = (const float*)d_in[0];
  const float* qw = (const float*)d_in[1];
  const float* kw = (const float*)d_in[2];
  const float* vw = (const float*)d_in[3];
  const float* ow = (const float*)d_in[4];
  const float* sp = (const float*)d_in[5];
  float* out = (float*)d_out;

  const size_t E = (size_t)SEQ * HID;  // 4M elements per matrix
  char* p = (char*)d_ws;
  bf16* hsb = (bf16*)p; p += E * 2;
  bf16* wqb = (bf16*)p; p += E * 2;
  bf16* wkb = (bf16*)p; p += E * 2;
  bf16* wvb = (bf16*)p; p += E * 2;
  bf16* wob = (bf16*)p; p += E * 2;
  bf16* Qb  = (bf16*)p; p += E * 2;
  bf16* Kb  = (bf16*)p; p += E * 2;
  bf16* Vb  = (bf16*)p; p += E * 2;
  bf16* Ab  = (bf16*)p; p += E * 2;
  unsigned char* maskb = (unsigned char*)p; p += NHEAD * NBLK * NBLK;
  p = (char*)(((uintptr_t)p + 255) & ~(uintptr_t)255);
  float* vbsum = (float*)p; p += (size_t)NHEAD * NBLK * HDIM * 4;

  const int cvtBlocks = (int)(E / 4 / 256);          // 4096
  cvt_bf16_kernel<<<cvtBlocks, 256, 0, stream>>>(hs, hsb, (int)E);
  cvt_bf16_kernel<<<cvtBlocks, 256, 0, stream>>>(qw, wqb, (int)E);
  cvt_bf16_kernel<<<cvtBlocks, 256, 0, stream>>>(kw, wkb, (int)E);
  cvt_bf16_kernel<<<cvtBlocks, 256, 0, stream>>>(vw, wvb, (int)E);
  cvt_bf16_kernel<<<cvtBlocks, 256, 0, stream>>>(ow, wob, (int)E);

  dim3 ggrid(HID / 128, SEQ / 128);                  // 16 x 16
  gemm_bf16_kernel<false><<<ggrid, 256, 0, stream>>>(hsb, wqb, (void*)Qb, SEQ, HID, HID);
  gemm_bf16_kernel<false><<<ggrid, 256, 0, stream>>>(hsb, wkb, (void*)Kb, SEQ, HID, HID);
  gemm_bf16_kernel<false><<<ggrid, 256, 0, stream>>>(hsb, wvb, (void*)Vb, SEQ, HID, HID);

  rope_kernel<<<(SEQ * NHEAD * 64) / 256, 256, 0, stream>>>(Qb, Kb);

  mask_kernel<<<NHEAD, 256, 0, stream>>>(sp, maskb);
  vbsum_kernel<<<NHEAD * NBLK, 128, 0, stream>>>(Vb, vbsum);

  dim3 agrid(NBLK, NHEAD);                           // 32 x 16
  attn_kernel<<<agrid, 256, 0, stream>>>(Qb, Kb, Vb, maskb, vbsum, Ab);

  gemm_bf16_kernel<true><<<ggrid, 256, 0, stream>>>(Ab, wob, (void*)out, SEQ, HID, HID);
}
